// multihead_self_attention_68083821576730
// MI455X (gfx1250) — compile-verified, matrix operand order checked by reading, not run on hardware
//
#include <hip/hip_runtime.h>
#include <hip/hip_bf16.h>

typedef __attribute__((ext_vector_type(16))) _Float16 v16h;
typedef __attribute__((ext_vector_type(8)))  _Float16 v8h;
typedef __attribute__((ext_vector_type(8)))  float    v8f;
typedef int v4i_ __attribute__((vector_size(4 * sizeof(int))));

#define B_  4
#define L_  2048
#define D_  1024
#define H_  16
#define DK_ 64
#define LN_THETA_OVER_DK 0.14391156642f  // ln(10000)/64

// ---------- optional CDNA5 async global->LDS path ----------
#if defined(__has_builtin)
#  if __has_builtin(__builtin_amdgcn_global_load_async_to_lds_b128) && \
      __has_builtin(__builtin_amdgcn_s_wait_asynccnt)
#    define USE_ASYNC_COPY 1
#  endif
#endif

__device__ __forceinline__ void copy16_g2lds(const _Float16* g, _Float16* l) {
#ifdef USE_ASYNC_COPY
    __builtin_amdgcn_global_load_async_to_lds_b128(
        (__attribute__((address_space(1))) v4i_*)(unsigned long long)g,
        (__attribute__((address_space(3))) v4i_*)(unsigned int)(unsigned long long)l,
        0, 0);
#else
    *(v8h*)l = *(const v8h*)g;
#endif
}
__device__ __forceinline__ void async_copy_wait() {
#ifdef USE_ASYNC_COPY
    __builtin_amdgcn_s_wait_asynccnt(0);
#endif
}

// ---------- WMMA helpers ----------
__device__ __forceinline__ v8f wmma16x16x32(v16h a, v16h b, v8f c) {
    return __builtin_amdgcn_wmma_f32_16x16x32_f16(
        false, a, false, b, (short)0, c, false, false);
}

// A/B fragment (16x32 f16, two 8-wide K runs per lane), f16 row-major source
// (global or LDS). lane l: row r = l&15, half = l>>4;
// elems 0..7 -> K = kb + half*8 + e, elems 8..15 -> K = kb+16+half*8+e.
__device__ __forceinline__ v16h load_frag_f16(const _Float16* base, int ld, int kb) {
    const int lane = threadIdx.x & 31;
    const int r = lane & 15, hf = lane >> 4;
    const _Float16* p = base + (size_t)r * ld + kb + hf * 8;
    v8h a = *(const v8h*)p;
    v8h b = *(const v8h*)(p + 16);
    v16h f;
#pragma unroll
    for (int e = 0; e < 8; ++e) { f[e] = a[e]; f[8 + e] = b[e]; }
    return f;
}

// Row reductions across the 16 lanes of a half-wave (C-tile row striping).
__device__ __forceinline__ float rowmax16(float x) {
#pragma unroll
    for (int off = 1; off < 16; off <<= 1) x = fmaxf(x, __shfl_xor(x, off, 32));
    return x;
}
__device__ __forceinline__ float rowsum16(float x) {
#pragma unroll
    for (int off = 1; off < 16; off <<= 1) x += __shfl_xor(x, off, 32);
    return x;
}

// ---------- Kernel 0: one-time f32 -> f16 conversion ----------
__global__ void __launch_bounds__(256)
cvt_f32_f16_kernel(const float* __restrict__ src, _Float16* __restrict__ dst, int n8) {
    const int i = blockIdx.x * blockDim.x + threadIdx.x;
    if (i >= n8) return;
    const float4* s = (const float4*)src + (size_t)i * 2;
    float4 a = s[0], b = s[1];
    v8h o;
    o[0]=(_Float16)a.x; o[1]=(_Float16)a.y; o[2]=(_Float16)a.z; o[3]=(_Float16)a.w;
    o[4]=(_Float16)b.x; o[5]=(_Float16)b.y; o[6]=(_Float16)b.z; o[7]=(_Float16)b.w;
    *(v8h*)(dst + (size_t)i * 8) = o;
}

// ---------- Kernel 1: QKV GEMM (f16) + fused RoPE ----------
// Wave tile: 32(M) x 64(N); a 64-wide N tile == one (which, head) pair.
// Q,K -> f16 [b][h][L][64] (RoPE applied). V -> f16 [b][h][64][L] (transposed).
__global__ void __launch_bounds__(256)
qkv_rope_kernel(const _Float16* __restrict__ Xh,
                const _Float16* __restrict__ Wq,
                const _Float16* __restrict__ Wk,
                const _Float16* __restrict__ Wv,
                const int*      __restrict__ tpos,
                _Float16* __restrict__ Qh,
                _Float16* __restrict__ Kh,
                _Float16* __restrict__ Vt) {
    const int wid  = threadIdx.x >> 5;
    const int tile = blockIdx.x * 8 + wid;     // 12288 wave-tiles
    const int nt = tile % 48;                  // 48 N-tiles of 64
    const int mt = tile / 48;                  // 256 M-tiles of 32
    const int lane = threadIdx.x & 31;
    const int col  = lane & 15, half = lane >> 4;

    const int m0 = mt * 32;
    const int b  = m0 >> 11;
    const int l0 = m0 & (L_ - 1);
    const int t0 = nt * 64;
    const int which = t0 >> 10;                // 0=Q 1=K 2=V
    const int h  = (t0 & 1023) >> 6;
    const int wr = t0 & 1023;
    const _Float16* W = (which == 0) ? Wq : (which == 1) ? Wk : Wv;

    v8f acc[2][4] = {};
    const _Float16* A0 = Xh + (size_t)m0 * D_;
    const _Float16* A1 = A0 + (size_t)16 * D_;
#pragma unroll 2
    for (int kb = 0; kb < D_; kb += 32) {
        v16h a0 = load_frag_f16(A0, D_, kb);
        v16h a1 = load_frag_f16(A1, D_, kb);
        if (kb + 32 < D_)
            __builtin_prefetch(A0 + (size_t)lane * D_ + kb + 32, 0, 1);
#pragma unroll
        for (int c = 0; c < 4; ++c) {
            v16h bb = load_frag_f16(W + (size_t)(wr + c * 16) * D_, D_, kb);
            acc[0][c] = wmma16x16x32(a0, bb, acc[0][c]);
            acc[1][c] = wmma16x16x32(a1, bb, acc[1][c]);
        }
    }

#pragma unroll
    for (int s = 0; s < 2; ++s) {
        const int ls = l0 + s * 16;
        if (which < 2) {
            _Float16* dst = ((which == 0) ? Qh : Kh) + (((size_t)b * H_ + h) * L_) * DK_;
#pragma unroll
            for (int c = 0; c < 4; ++c) {
                const int d = c * 16 + col;
                const float inv = __expf(-(float)(d & ~1) * LN_THETA_OVER_DK);
#pragma unroll
                for (int v = 0; v < 8; ++v) {
                    const int row = ls + v + 8 * half;
                    const float pos = (float)tpos[row];
                    const float ang = pos * inv;
                    const float cs = __cosf(ang), sn = __sinf(ang);
                    const float val = acc[s][c][v];
                    const float other = __shfl_xor(val, 1, 32);
                    const float r = (d & 1) ? (other * sn + val * cs)
                                            : (val * cs - other * sn);
                    dst[(size_t)row * DK_ + d] = (_Float16)r;
                }
            }
        } else {
            _Float16* dst = Vt + (((size_t)b * H_ + h) * DK_) * L_;
#pragma unroll
            for (int c = 0; c < 4; ++c) {
                const int d = c * 16 + col;
                v8h pk;
#pragma unroll
                for (int v = 0; v < 8; ++v) pk[v] = (_Float16)acc[s][c][v];
                *(v8h*)(dst + (size_t)d * L_ + ls + 8 * half) = pk;
            }
        }
    }
}

// ---------- Kernel 2: causal flash attention ----------
// 4 waves / workgroup share one (b,h) and 64 consecutive Q rows; each 32-row
// KV chunk is staged once into LDS (async when available) and reused by all
// 4 waves. Per-wave causal chunk skipping (wave-uniform, EXEC stays full).
__global__ void __launch_bounds__(128)
flash_attn_kernel(const _Float16* __restrict__ Qh,
                  const _Float16* __restrict__ Kh,
                  const _Float16* __restrict__ Vt,
                  _Float16* __restrict__ At) {
    __shared__ __align__(16) _Float16 Kl[32 * 64];     // [kv_local][d]
    __shared__ __align__(16) _Float16 Vl[64 * 32];     // [d][kv_local]
    __shared__ __align__(16) _Float16 Pl[4][16 * 32];  // per-wave P

    const int bid = blockIdx.x;               // B*H*(L/64) = 2048
    const int qb  = bid & 31;
    const int bh  = bid >> 5;
    const int h   = bh & (H_ - 1);
    const int b   = bh >> 4;
    const int qbase = qb * 64;

    const int w    = threadIdx.x >> 5;
    const int lane = threadIdx.x & 31;
    const int col  = lane & 15, half = lane >> 4;
    const int q0   = qbase + w * 16;

    const _Float16* Qbh = Qh + (((size_t)b * H_ + h) * L_) * DK_;
    const _Float16* Kbh = Kh + (((size_t)b * H_ + h) * L_) * DK_;
    const _Float16* Vbh = Vt + (((size_t)b * H_ + h) * DK_) * L_;

    const v16h qa0 = load_frag_f16(Qbh + (size_t)q0 * DK_, DK_, 0);
    const v16h qa1 = load_frag_f16(Qbh + (size_t)q0 * DK_, DK_, 32);

    v8f o[4] = {};
    float mrun[8], srun[8];
#pragma unroll
    for (int v = 0; v < 8; ++v) { mrun[v] = -1.0e30f; srun[v] = 0.0f; }

    const int nchunks = 2 * qb + 2;           // covers rows up to qbase+63
    for (int kc = 0; kc < nchunks; ++kc) {
        const int kv0 = kc * 32;

        // ---- stage K (32x64) and V (64x32) tiles into LDS ----
        {
            const int t = threadIdx.x;
#pragma unroll
            for (int i = 0; i < 2; ++i) {     // K: 256 x 16B chunks
                const int cc = t + i * 128;
                const int r = cc >> 3, off = (cc & 7) * 8;
                copy16_g2lds(Kbh + (size_t)(kv0 + r) * DK_ + off, &Kl[r * 64 + off]);
            }
#pragma unroll
            for (int i = 0; i < 2; ++i) {     // V: 256 x 16B chunks
                const int cc = t + i * 128;
                const int d = cc >> 2, off = (cc & 3) * 8;
                copy16_g2lds(Vbh + (size_t)d * L_ + kv0 + off, &Vl[d * 32 + off]);
            }
        }
        async_copy_wait();
        __syncthreads();

        if (kv0 <= q0 + 15) {                 // wave-uniform causal skip
            v8f s0 = {}, s1 = {};
            {
                v16h b0a = load_frag_f16(Kl, 64, 0);
                v16h b0b = load_frag_f16(Kl, 64, 32);
                v16h b1a = load_frag_f16(Kl + 16 * 64, 64, 0);
                v16h b1b = load_frag_f16(Kl + 16 * 64, 64, 32);
                s0 = wmma16x16x32(qa0, b0a, s0);
                s0 = wmma16x16x32(qa1, b0b, s0);
                s1 = wmma16x16x32(qa0, b1a, s1);
                s1 = wmma16x16x32(qa1, b1b, s1);
            }

            float alpha[8];
#pragma unroll
            for (int v = 0; v < 8; ++v) {
                const int row = q0 + v + 8 * half;
                const int c0 = kv0 + col, c1 = c0 + 16;
                float x0 = s0[v] * 0.125f; if (c0 > row) x0 = -3.0e38f;
                float x1 = s1[v] * 0.125f; if (c1 > row) x1 = -3.0e38f;
                const float mx = rowmax16(fmaxf(x0, x1));
                const float mn = fmaxf(mrun[v], mx);
                const float al = __expf(mrun[v] - mn);
                const float e0 = __expf(x0 - mn);
                const float e1 = __expf(x1 - mn);
                srun[v] = srun[v] * al + rowsum16(e0 + e1);
                mrun[v] = mn;
                alpha[v] = al;
                Pl[w][(v + 8 * half) * 32 + col]      = (_Float16)e0;
                Pl[w][(v + 8 * half) * 32 + 16 + col] = (_Float16)e1;
            }
#pragma unroll
            for (int c = 0; c < 4; ++c)
#pragma unroll
                for (int v = 0; v < 8; ++v) o[c][v] *= alpha[v];

            // same-wave LDS ops are in-order: P stores -> P fragment load
            const v16h pa = load_frag_f16(Pl[w], 32, 0);
#pragma unroll
            for (int c = 0; c < 4; ++c) {
                v16h vb = load_frag_f16(Vl + (size_t)(c * 16) * 32, 32, 0);
                o[c] = wmma16x16x32(pa, vb, o[c]);
            }
        }
        __syncthreads();                      // protect Kl/Vl before restage
    }

    // normalize + store attn out: f16 [b][l][h*64+d]
#pragma unroll
    for (int c = 0; c < 4; ++c) {
#pragma unroll
        for (int v = 0; v < 8; ++v) {
            const int row = q0 + v + 8 * half;
            const float val = o[c][v] / fmaxf(srun[v], 1e-30f);
            At[((size_t)b * L_ + row) * D_ + h * DK_ + c * 16 + col] = (_Float16)val;
        }
    }
}

// ---------- Kernel 3: output projection ----------
// out[m][n] = sum_k At[m][k] * Wo[n][k];  M=8192, N=K=1024. 32x64 wave tiles.
__global__ void __launch_bounds__(256)
oproj_kernel(const _Float16* __restrict__ At,
             const _Float16* __restrict__ Woh,
             float* __restrict__ out) {
    const int wid  = threadIdx.x >> 5;
    const int tile = blockIdx.x * 8 + wid;    // 4096 wave-tiles
    const int nt = tile & 15;                 // 16 N-tiles of 64
    const int mt = tile >> 4;                 // 256 M-tiles of 32
    const int lane = threadIdx.x & 31;
    const int col  = lane & 15, half = lane >> 4;

    const int m0 = mt * 32;
    const int n0 = nt * 64;

    v8f acc[2][4] = {};
    const _Float16* A0 = At + (size_t)m0 * D_;
    const _Float16* A1 = A0 + (size_t)16 * D_;
#pragma unroll 2
    for (int kb = 0; kb < D_; kb += 32) {
        v16h a0 = load_frag_f16(A0, D_, kb);
        v16h a1 = load_frag_f16(A1, D_, kb);
        if (kb + 32 < D_)
            __builtin_prefetch(A0 + (size_t)lane * D_ + kb + 32, 0, 1);
#pragma unroll
        for (int c = 0; c < 4; ++c) {
            v16h bb = load_frag_f16(Woh + (size_t)(n0 + c * 16) * D_, D_, kb);
            acc[0][c] = wmma16x16x32(a0, bb, acc[0][c]);
            acc[1][c] = wmma16x16x32(a1, bb, acc[1][c]);
        }
    }
#pragma unroll
    for (int s = 0; s < 2; ++s)
#pragma unroll
        for (int c = 0; c < 4; ++c)
#pragma unroll
            for (int v = 0; v < 8; ++v)
                out[(size_t)(m0 + s * 16 + v + 8 * half) * D_ + n0 + c * 16 + col] =
                    acc[s][c][v];
}

// ---------- launcher ----------
extern "C" void kernel_launch(void* const* d_in, const int* in_sizes, int n_in,
                              void* d_out, int out_size, void* d_ws, size_t ws_size,
                              hipStream_t stream) {
    const float* x  = (const float*)d_in[0];
    const float* wq = (const float*)d_in[1];
    const float* wk = (const float*)d_in[2];
    const float* wv = (const float*)d_in[3];
    const float* wo = (const float*)d_in[4];
    const int*   tp = (const int*)d_in[5];
    float* out = (float*)d_out;

    const size_t xe = (size_t)B_ * L_ * D_;   // 8,388,608
    const size_t we = (size_t)D_ * D_;        // 1,048,576

    _Float16* Xh  = (_Float16*)d_ws;          // [8192][1024]
    _Float16* Wqh = Xh  + xe;
    _Float16* Wkh = Wqh + we;
    _Float16* Wvh = Wkh + we;
    _Float16* Woh = Wvh + we;
    _Float16* Qh  = Woh + we;                 // [b][h][L][64]
    _Float16* Kh  = Qh  + xe;                 // [b][h][L][64]
    _Float16* Vt  = Kh  + xe;                 // [b][h][64][L]
    _Float16* At  = Vt  + xe;                 // [b][l][1024]

    // 0) one-time f32 -> f16 conversions (bandwidth-bound)
    cvt_f32_f16_kernel<<<(int)(xe / 8 / 256), 256, 0, stream>>>(x,  Xh,  (int)(xe / 8));
    cvt_f32_f16_kernel<<<(int)(we / 8 / 256), 256, 0, stream>>>(wq, Wqh, (int)(we / 8));
    cvt_f32_f16_kernel<<<(int)(we / 8 / 256), 256, 0, stream>>>(wk, Wkh, (int)(we / 8));
    cvt_f32_f16_kernel<<<(int)(we / 8 / 256), 256, 0, stream>>>(wv, Wvh, (int)(we / 8));
    cvt_f32_f16_kernel<<<(int)(we / 8 / 256), 256, 0, stream>>>(wo, Woh, (int)(we / 8));

    // 1) QKV projection + RoPE + layout conversion
    qkv_rope_kernel<<<1536, 256, 0, stream>>>(Xh, Wqh, Wkh, Wvh, tp, Qh, Kh, Vt);
    // 2) causal flash attention (LDS-shared KV tiles, async staging)
    flash_attn_kernel<<<B_ * H_ * (L_ / 64), 128, 0, stream>>>(Qh, Kh, Vt, At);
    // 3) output projection
    oproj_kernel<<<512, 256, 0, stream>>>(At, Woh, out);
}
